// RingDilatedAttentionFixedSimple_64690797412684
// MI455X (gfx1250) — matrix operand order repaired and checked by reading, not compiled, not run on hardware
//
#include <hip/hip_runtime.h>
#include <hip/hip_bf16.h>

typedef __bf16 bf16;
typedef __attribute__((ext_vector_type(16))) __bf16 v16bf;
typedef __attribute__((ext_vector_type(8)))  __bf16 bf16x8;
typedef __attribute__((ext_vector_type(8)))  float  v8f;

union FragBF { v16bf v; bf16x8 h[2]; };

#define WMMA_BF16(A, B, C) \
  __builtin_amdgcn_wmma_f32_16x16x32_bf16(false, (A), false, (B), (short)0, (C), false, false)

// Async global->LDS copy (CDNA5, ASYNCcnt-tracked). %0 = LDS byte address
// (low 32 bits of flat == LDS offset per aperture mapping), %1 = 64-bit VA.
#define ASYNC_CP_B128(lds32, gaddr)                                         \
  asm volatile("global_load_async_to_lds_b128 %0, %1, off"                  \
               :: "v"(lds32), "v"(gaddr) : "memory")
#define ASYNC_CP_B32(lds32, gaddr)                                          \
  asm volatile("global_load_async_to_lds_b32 %0, %1, off"                   \
               :: "v"(lds32), "v"(gaddr) : "memory")
#define ASYNC_WAIT(imm) asm volatile("s_wait_asynccnt " #imm ::: "memory")

__device__ __forceinline__ unsigned lds_addr32(const void* p) {
  return (unsigned)(unsigned long long)p;  // flat addr low 32 bits = LDS offset
}

__device__ __forceinline__ bf16x8 cvt8(const float* __restrict__ p) {
  bf16x8 r;
#pragma unroll
  for (int i = 0; i < 8; ++i) r[i] = (bf16)p[i];
  return r;
}

constexpr int Bb = 2, S = 2048, D = 1024, H = 16, HD = 64;
constexpr int LP = 36;           // LDS row pitch in f32 (144 B, 16 B aligned)
constexpr float SCALE = 0.125f;  // 1/sqrt(64)

// ---------------------------------------------------------------------------
// Kernel 1: QKV projection  C[4096,3072] = x[4096,1024] @ Wqkv[1024,3072] + b
// 64x64 tile/block, 8 waves. Tiles staged in f32 via double-buffered
// global_load_async_to_lds (B transposed during the async scatter), converted
// to bf16 while assembling WMMA fragments. Epilogue scatters q/k/v:
//   qb,kb : [b*H+h][s][64]  (Q pre-scaled by 1/sqrt(hd))
//   vb    : [b*H+h][64][s]  (transposed -> contiguous V B-fragments later)
// ---------------------------------------------------------------------------
__global__ __launch_bounds__(256)
void qkv_proj_kernel(const float* __restrict__ x, const float* __restrict__ Wqkv,
                     const float* __restrict__ bqkv,
                     bf16* __restrict__ qb, bf16* __restrict__ kb, bf16* __restrict__ vb) {
  __shared__ __align__(16) float As[2][64 * LP];  // [m][k]
  __shared__ __align__(16) float Bs[2][64 * LP];  // [n][k] (transposed)

  const int n0 = blockIdx.x * 64;
  const int m0 = blockIdx.y * 64;
  const int t    = threadIdx.x;
  const int wave = t >> 5, lane = t & 31, ln = lane & 15, half = lane >> 4;
  const int wm = wave & 3, wn = wave >> 2;

  v8f acc0 = {0.f,0.f,0.f,0.f,0.f,0.f,0.f,0.f};
  v8f acc1 = acc0;

  const int ar = t >> 2, ac = (t & 3) * 8;   // A staging coords (8 contiguous)
  const int bn = t & 63, bk = (t >> 6) * 8;  // B staging coords (transpose)

  auto issue_tile = [&](int k0, int buf) {
    // x tile 64x32: 2 async b128 per thread, natural layout
    const float* xs = x + (size_t)(m0 + ar) * D + k0 + ac;
    const unsigned la = lds_addr32(&As[buf][ar * LP + ac]);
    ASYNC_CP_B128(la,      (unsigned long long)xs);
    ASYNC_CP_B128(la + 16, (unsigned long long)(xs + 4));
    // Wqkv tile 32x64 transposed into [n][k]: 8 async b32 per thread
#pragma unroll
    for (int i = 0; i < 8; ++i) {
      const float* wq = Wqkv + (size_t)(k0 + bk + i) * (3 * D) + n0 + bn;
      ASYNC_CP_B32(lds_addr32(&Bs[buf][bn * LP + bk + i]),
                   (unsigned long long)wq);
    }
  };

  issue_tile(0, 0);
  for (int kk = 0; kk < D / 32; ++kk) {
    const int cur = kk & 1;
    if (kk + 1 < D / 32) {
      issue_tile((kk + 1) * 32, cur ^ 1);
      ASYNC_WAIT(10);  // 10 just issued for next tile -> previous tile done
    } else {
      ASYNC_WAIT(0);
    }
    __syncthreads();

    FragBF a, b0, b1;
    const float* arow = &As[cur][(wm * 16 + ln) * LP];
    a.h[0] = cvt8(arow + half * 16);
    a.h[1] = cvt8(arow + half * 16 + 8);
    const float* br0 = &Bs[cur][(wn * 32 + ln) * LP + half * 16];
    b0.h[0] = cvt8(br0);
    b0.h[1] = cvt8(br0 + 8);
    const float* br1 = &Bs[cur][(wn * 32 + 16 + ln) * LP + half * 16];
    b1.h[0] = cvt8(br1);
    b1.h[1] = cvt8(br1 + 8);

    acc0 = WMMA_BF16(a.v, b0.v, acc0);
    acc1 = WMMA_BF16(a.v, b1.v, acc1);
    __syncthreads();  // protect buffer about to be refilled
  }

#pragma unroll
  for (int sub = 0; sub < 2; ++sub) {
    const v8f acc = sub ? acc1 : acc0;
    const int col   = n0 + wn * 32 + sub * 16 + ln;  // [0, 3072)
    const int which = col >> 10;                     // 0=q 1=k 2=v
    const int h = (col >> 6) & 15;
    const int d = col & 63;
    const float bias = bqkv[col];
#pragma unroll
    for (int v = 0; v < 8; ++v) {
      const int m = m0 + wm * 16 + half * 8 + v;     // [0, 4096)
      const int b = m >> 11, s = m & 2047;
      const int bh = b * H + h;
      const float val = acc[v] + bias;
      if (which == 0)      qb[((size_t)bh * S + s) * HD + d] = (bf16)(val * SCALE);
      else if (which == 1) kb[((size_t)bh * S + s) * HD + d] = (bf16)val;
      else                 vb[((size_t)bh * HD + d) * S + s] = (bf16)val;
    }
  }
}

// ---------------------------------------------------------------------------
// Kernel 2: flash attention per (b,h). 4 waves/block, one 16-query strip per
// wave, 32-key inner tiles. Q@K^T (4 WMMAs), online softmax via __shfl_xor,
// P relaid C->A through per-wave LDS, P@V (4 WMMAs from transposed V).
// Output bf16 into ab[b*S+s][h*64+d] for the out-projection.
// ---------------------------------------------------------------------------
__global__ __launch_bounds__(128)
void attn_kernel(const bf16* __restrict__ qb, const bf16* __restrict__ kb,
                 const bf16* __restrict__ vb, bf16* __restrict__ ab) {
  __shared__ __align__(16) bf16 Plds[4][16][32];

  const int tid  = threadIdx.x;
  const int wave = tid >> 5, lane = tid & 31, ln = lane & 15, half = lane >> 4;
  const int bh = blockIdx.y;
  const int b = bh >> 4, h = bh & 15;
  const int q0 = blockIdx.x * 64 + wave * 16;

  const bf16* Q  = qb + (size_t)bh * S * HD;
  const bf16* K  = kb + (size_t)bh * S * HD;
  const bf16* Vt = vb + (size_t)bh * HD * S;  // [d][s]

  FragBF a0, a1;  // Q strip 16x64 = two 16x32 A fragments
  const bf16* qrow = Q + (size_t)(q0 + ln) * HD;
  a0.h[0] = *(const bf16x8*)(qrow + half * 16);
  a0.h[1] = *(const bf16x8*)(qrow + half * 16 + 8);
  a1.h[0] = *(const bf16x8*)(qrow + 32 + half * 16);
  a1.h[1] = *(const bf16x8*)(qrow + 40 + half * 16);

  v8f o0 = {0.f,0.f,0.f,0.f,0.f,0.f,0.f,0.f};
  v8f o1 = o0, o2 = o0, o3 = o0;
  float mrow[8], lrow[8];
#pragma unroll
  for (int v = 0; v < 8; ++v) { mrow[v] = -3.0e38f; lrow[v] = 0.f; }

  for (int j = 0; j < S / 32; ++j) {
    const int key0 = j * 32;
    if (j + 1 < S / 32)
      __builtin_prefetch(K + (size_t)(key0 + 32 + ln) * HD, 0, 0);

    v8f s0 = {0.f,0.f,0.f,0.f,0.f,0.f,0.f,0.f};
    v8f s1 = s0;
    {
      FragBF bk;
      const bf16* kr = K + (size_t)(key0 + ln) * HD + half * 16;
      bk.h[0] = *(const bf16x8*)(kr);      bk.h[1] = *(const bf16x8*)(kr + 8);
      s0 = WMMA_BF16(a0.v, bk.v, s0);
      bk.h[0] = *(const bf16x8*)(kr + 32); bk.h[1] = *(const bf16x8*)(kr + 40);
      s0 = WMMA_BF16(a1.v, bk.v, s0);
      kr = K + (size_t)(key0 + 16 + ln) * HD + half * 16;
      bk.h[0] = *(const bf16x8*)(kr);      bk.h[1] = *(const bf16x8*)(kr + 8);
      s1 = WMMA_BF16(a0.v, bk.v, s1);
      bk.h[0] = *(const bf16x8*)(kr + 32); bk.h[1] = *(const bf16x8*)(kr + 40);
      s1 = WMMA_BF16(a1.v, bk.v, s1);
    }

    // online softmax over this tile's 32 keys (row = half*8 + v)
#pragma unroll
    for (int v = 0; v < 8; ++v) {
      float tmax = fmaxf(s0[v], s1[v]);
#pragma unroll
      for (int mk = 1; mk <= 8; mk <<= 1) tmax = fmaxf(tmax, __shfl_xor(tmax, mk, 32));
      const float mnew  = fmaxf(mrow[v], tmax);
      const float alpha = __expf(mrow[v] - mnew);
      mrow[v] = mnew;
      const float p0 = __expf(s0[v] - mnew);
      const float p1 = __expf(s1[v] - mnew);
      float rs = p0 + p1;
#pragma unroll
      for (int mk = 1; mk <= 8; mk <<= 1) rs += __shfl_xor(rs, mk, 32);
      lrow[v] = lrow[v] * alpha + rs;
      o0[v] *= alpha; o1[v] *= alpha; o2[v] *= alpha; o3[v] *= alpha;
      Plds[wave][half * 8 + v][ln]      = (bf16)p0;
      Plds[wave][half * 8 + v][16 + ln] = (bf16)p1;
    }
    __syncthreads();

    FragBF pa;  // P as a 16x32 A fragment
    const bf16* prow = &Plds[wave][ln][0];
    pa.h[0] = *(const bf16x8*)(prow + half * 16);
    pa.h[1] = *(const bf16x8*)(prow + half * 16 + 8);

#pragma unroll
    for (int c = 0; c < 4; ++c) {
      FragBF bv;  // V^T rows contiguous in s -> contiguous B fragment
      const bf16* vr = Vt + (size_t)(c * 16 + ln) * S + key0 + half * 16;
      bv.h[0] = *(const bf16x8*)(vr);
      bv.h[1] = *(const bf16x8*)(vr + 8);
      v8f& o = (c == 0) ? o0 : (c == 1) ? o1 : (c == 2) ? o2 : o3;
      o = WMMA_BF16(pa.v, bv.v, o);
    }
    __syncthreads();
  }

#pragma unroll
  for (int v = 0; v < 8; ++v) {
    const float inv = 1.0f / lrow[v];
    const int s = q0 + half * 8 + v;
    bf16* orow = ab + ((size_t)(b * S + s)) * D + h * HD;
    orow[ln]      = (bf16)(o0[v] * inv);
    orow[16 + ln] = (bf16)(o1[v] * inv);
    orow[32 + ln] = (bf16)(o2[v] * inv);
    orow[48 + ln] = (bf16)(o3[v] * inv);
  }
}

// ---------------------------------------------------------------------------
// Kernel 3: output projection  out[4096,1024] = ab(bf16) @ Wout + bout (f32)
// B tile async-staged (transposed f32, double-buffered); A fragments loaded
// directly from the bf16 workspace.
// ---------------------------------------------------------------------------
__global__ __launch_bounds__(256)
void out_proj_kernel(const bf16* __restrict__ ab, const float* __restrict__ Wout,
                     const float* __restrict__ bout, float* __restrict__ out) {
  __shared__ __align__(16) float Bs[2][64 * LP];  // Wout tile transposed [n][k]

  const int n0 = blockIdx.x * 64;
  const int m0 = blockIdx.y * 64;
  const int t    = threadIdx.x;
  const int wave = t >> 5, lane = t & 31, ln = lane & 15, half = lane >> 4;
  const int wm = wave & 3, wn = wave >> 2;
  const int bn = t & 63, bk = (t >> 6) * 8;

  v8f acc0 = {0.f,0.f,0.f,0.f,0.f,0.f,0.f,0.f};
  v8f acc1 = acc0;

  auto issue_tile = [&](int k0, int buf) {
#pragma unroll
    for (int i = 0; i < 8; ++i) {
      const float* wp = Wout + (size_t)(k0 + bk + i) * D + n0 + bn;
      ASYNC_CP_B32(lds_addr32(&Bs[buf][bn * LP + bk + i]),
                   (unsigned long long)wp);
    }
  };

  issue_tile(0, 0);
  for (int kk = 0; kk < D / 32; ++kk) {
    const int cur = kk & 1;
    const int k0 = kk * 32;
    if (kk + 1 < D / 32) {
      issue_tile((kk + 1) * 32, cur ^ 1);
      ASYNC_WAIT(8);
    } else {
      ASYNC_WAIT(0);
    }
    __syncthreads();

    FragBF a, b0, b1;
    const bf16* arow = ab + (size_t)(m0 + wm * 16 + ln) * D + k0;
    a.h[0] = *(const bf16x8*)(arow + half * 16);
    a.h[1] = *(const bf16x8*)(arow + half * 16 + 8);
    const float* br0 = &Bs[cur][(wn * 32 + ln) * LP + half * 16];
    b0.h[0] = cvt8(br0);
    b0.h[1] = cvt8(br0 + 8);
    const float* br1 = &Bs[cur][(wn * 32 + 16 + ln) * LP + half * 16];
    b1.h[0] = cvt8(br1);
    b1.h[1] = cvt8(br1 + 8);

    acc0 = WMMA_BF16(a.v, b0.v, acc0);
    acc1 = WMMA_BF16(a.v, b1.v, acc1);
    __syncthreads();
  }

#pragma unroll
  for (int sub = 0; sub < 2; ++sub) {
    const v8f acc = sub ? acc1 : acc0;
    const int col = n0 + wn * 32 + sub * 16 + ln;
    const float bias = bout[col];
#pragma unroll
    for (int v = 0; v < 8; ++v) {
      const int m = m0 + wm * 16 + half * 8 + v;
      out[(size_t)m * D + col] = acc[v] + bias;
    }
  }
}

// ---------------------------------------------------------------------------
extern "C" void kernel_launch(void* const* d_in, const int* in_sizes, int n_in,
                              void* d_out, int out_size, void* d_ws, size_t ws_size,
                              hipStream_t stream) {
  const float* x    = (const float*)d_in[0];
  const float* Wqkv = (const float*)d_in[1];
  const float* bqkv = (const float*)d_in[2];
  const float* Wout = (const float*)d_in[3];
  const float* bout = (const float*)d_in[4];
  float* out = (float*)d_out;

  const size_t per = (size_t)Bb * H * S * HD;  // 4,194,304 bf16 elems = 8 MB
  bf16* qb = (bf16*)d_ws;
  bf16* kb = qb + per;
  bf16* vb = kb + per;
  bf16* ab = vb + per;                         // [B*S][D] bf16, 8 MB

  qkv_proj_kernel<<<dim3(3 * D / 64, Bb * S / 64), dim3(256), 0, stream>>>(
      x, Wqkv, bqkv, qb, kb, vb);
  attn_kernel<<<dim3(S / 64, Bb * H), dim3(128), 0, stream>>>(qb, kb, vb, ab);
  out_proj_kernel<<<dim3(D / 64, Bb * S / 64), dim3(256), 0, stream>>>(
      ab, Wout, bout, out);
}
